// SAPA_38809324487107
// MI455X (gfx1250) — matrix operand order repaired and checked by reading, not compiled, not run on hardware
//
#include <hip/hip_runtime.h>
#include <hip/hip_bf16.h>

// ---------------- problem constants (from reference) ----------------
constexpr int kB      = 4;
constexpr int kC      = 256;   // channels of y and x
constexpr int kE      = 32;    // embedding dim
constexpr int kH      = 64;    // low-res H (x)
constexpr int kW      = 64;    // low-res W (x)
constexpr int kHY     = 128;   // hi-res H (y)
constexpr int kWY     = 128;   // hi-res W (y)
constexpr int kGroups = 32;
constexpr int kCPG    = kC / kGroups; // 8 channels per group (contiguous in BCHW!)
constexpr int kKs     = 5;     // kernel size
constexpr int kPad    = 2;
constexpr float kEps  = 1e-5f;

typedef __attribute__((ext_vector_type(16))) __bf16 v16bf;
typedef __attribute__((ext_vector_type(8)))  float  v8f;
typedef __attribute__((ext_vector_type(4)))  float  v4f;

// ---------------- CDNA5 async global->LDS copy (guarded) ----------------
// Builtin signature (probe-confirmed by round-2 diagnostic):
//   void __builtin_amdgcn_global_load_async_to_lds_b128(
//        AS1 int4* gsrc, AS3 int4* ldst, imm int offset, imm int cpol)
typedef int i4v __attribute__((vector_size(16)));
typedef __attribute__((address_space(1))) i4v* gptr_i4;
typedef __attribute__((address_space(3))) i4v* lptr_i4;

#if defined(__gfx1250__) && __has_builtin(__builtin_amdgcn_global_load_async_to_lds_b128)
#define ASYNC_COPY16(gsrc, ldst)                                  \
  __builtin_amdgcn_global_load_async_to_lds_b128((gptr_i4)(gsrc), \
                                                 (lptr_i4)(ldst), 0, 0)
#if __has_builtin(__builtin_amdgcn_s_wait_asynccnt)
#define ASYNC_WAIT() __builtin_amdgcn_s_wait_asynccnt(0)
#else
#define ASYNC_WAIT() asm volatile("s_wait_asynccnt 0" ::: "memory")
#endif
#else
#define ASYNC_COPY16(gsrc, ldst) (*(v4f*)(ldst) = *(const v4f*)(gsrc))
#define ASYNC_WAIT() ((void)0)
#endif

// ==================================================================
// 1) GroupNorm statistics: one block per (b, group). Group data is a
//    contiguous run of kCPG*HW floats in BCHW layout -> coalesced.
// ==================================================================
template <int HW>
__global__ __launch_bounds__(256)
void gn_stats_kernel(const float* __restrict__ src,
                     float* __restrict__ stats /* (B*G, 2) */) {
  const int bg = blockIdx.x;                    // b*kGroups + g
  const long base = (long)bg * (kCPG * HW);
  const int n = kCPG * HW;

  float s = 0.f, s2 = 0.f;
  for (int i = threadIdx.x; i < n; i += blockDim.x) {
    float v = src[base + i];
    s  += v;
    s2  = fmaf(v, v, s2);
  }
  // intra-wave (wave32) reduction
  for (int off = 16; off > 0; off >>= 1) {
    s  += __shfl_down(s,  off);
    s2 += __shfl_down(s2, off);
  }
  __shared__ float ls[8], ls2[8];
  const int wid = threadIdx.x >> 5, lid = threadIdx.x & 31;
  if (lid == 0) { ls[wid] = s; ls2[wid] = s2; }
  __syncthreads();
  if (threadIdx.x == 0) {
    float a = 0.f, b2 = 0.f;
    for (int i = 0; i < 8; ++i) { a += ls[i]; b2 += ls2[i]; }
    const float mean = a / (float)n;
    const float var  = b2 / (float)n - mean * mean;
    stats[bg * 2 + 0] = mean;
    stats[bg * 2 + 1] = rsqrtf(var + kEps);
  }
}

// ==================================================================
// 2) Fold GN stats + affine into per-(b,c) scale/bias:
//    xn = x*gs + bs  with gs = rstd*gamma, bs = beta - mean*gs
// ==================================================================
__global__ __launch_bounds__(256)
void gn_fuse_kernel(const float* __restrict__ stats,
                    const float* __restrict__ gamma,
                    const float* __restrict__ beta,
                    float* __restrict__ gs,
                    float* __restrict__ bs) {
  const int i = blockIdx.x * blockDim.x + threadIdx.x; // over B*C
  if (i >= kB * kC) return;
  const int b = i / kC, c = i % kC, g = c / kCPG;
  const float mean = stats[(b * kGroups + g) * 2 + 0];
  const float rstd = stats[(b * kGroups + g) * 2 + 1];
  const float sc   = gamma[c] * rstd;
  gs[i] = sc;
  bs[i] = beta[c] - mean * sc;
}

// ==================================================================
// 3) WMMA projection:  dst(b,pos,E) = GN(src)(b,c,pos) @ Wmat(E,C)^T + bias
//    One wave computes a 16-position x 32-E tile via two
//    v_wmma_f32_16x16x32_bf16 accumulators, K-loop over 8 chunks of 32
//    channels. A/B fragments both use contiguous k = 16*half + j order.
// ==================================================================
template <int HWp>
__global__ __launch_bounds__(128)
void project_wmma_kernel(const float* __restrict__ src,   // (B,C,HWp)
                         const float* __restrict__ gs,    // (B,C)
                         const float* __restrict__ bs,    // (B,C)
                         const float* __restrict__ wmat,  // (E,C)
                         const float* __restrict__ bias,  // (E)
                         float* __restrict__ dst) {       // (B,HWp,E)
  const int wavesPerBlock = blockDim.x >> 5;
  const int wave = blockIdx.x * wavesPerBlock + (threadIdx.x >> 5);
  const int lane = threadIdx.x & 31;
  const long rowBase = (long)wave * 16;      // row index over B*HWp (HWp % 16 == 0)
  const int b  = (int)(rowBase / HWp);
  const int n0 = (int)(rowBase % HWp);
  const int r       = lane & 15;
  const int hiHalf  = lane >> 4;             // 0 or 1
  const int n       = n0 + r;                // this lane's A-row position

  const float* __restrict__ srcB = src + (long)b * kC * HWp;
  const float* __restrict__ gsB  = gs + b * kC;
  const float* __restrict__ bsB  = bs + b * kC;

  v8f acc0 = {};  // E columns 0..15
  v8f acc1 = {};  // E columns 16..31

  for (int kc = 0; kc < kC; kc += 32) {
    // ---- A fragment: GN applied on the fly, f32 -> bf16 ----
    v16bf a;
#pragma unroll
    for (int j = 0; j < 16; ++j) {
      const int kk = hiHalf * 16 + j;
      const int c  = kc + kk;
      const float v = fmaf(srcB[(long)c * HWp + n], gsB[c], bsB[c]);
      a[j] = (__bf16)v;
    }
    // ---- B fragments: contiguous 16-float runs of Wmat rows ----
    const float* __restrict__ w0 = wmat + (long)(lane & 15) * kC + kc + hiHalf * 16;
    const float* __restrict__ w1 = wmat + (long)((lane & 15) + 16) * kC + kc + hiHalf * 16;
    v16bf bf0, bf1;
#pragma unroll
    for (int j = 0; j < 16; ++j) {
      bf0[j] = (__bf16)w0[j];
      bf1[j] = (__bf16)w1[j];
    }
    acc0 = __builtin_amdgcn_wmma_f32_16x16x32_bf16(false, a, false, bf0,
                                                   (short)0, acc0, false, false);
    acc1 = __builtin_amdgcn_wmma_f32_16x16x32_bf16(false, a, false, bf1,
                                                   (short)0, acc1, false, false);
  }

  // ---- store: C/D layout -> lane L = col N=L%16, VGPR v = row M=v+8*(L>=16)
  const int e0 = lane & 15;
#pragma unroll
  for (int v = 0; v < 8; ++v) {
    const int pos = n0 + v + hiHalf * 8;
    float* __restrict__ dp = dst + ((long)b * HWp + pos) * kE;
    dp[e0]      = acc0[v] + bias[e0];
    dp[e0 + 16] = acc1[v] + bias[e0 + 16];
  }
}

// ==================================================================
// 4) Logits + softmax. One block per (b, low-res row h). k halo tile
//    (5 rows x 68 cols x 32 e) staged into LDS with async b128 copies
//    (all 16B-aligned since E=32); each thread owns one hi-res pixel,
//    computes 25 dot-32 logits + register softmax; attn written
//    offset-major for coalesced reads in the aggregation kernel.
// ==================================================================
__global__ __launch_bounds__(256)
void sapa_logits_kernel(const float* __restrict__ q,    // (B, 2H, 2W, E)
                        const float* __restrict__ k,    // (B, H, W, E)
                        float* __restrict__ attn) {     // (B*H, 25, 256)
  const int bh = blockIdx.x;
  const int b  = bh / kH;
  const int h  = bh % kH;

  __shared__ __align__(16) float kt[5 * 68 * kE];  // 43520 B
  constexpr int KT_VEC = 5 * 68 * (kE / 4);        // 2720 float4 transfers
  for (int t = threadIdx.x; t < KT_VEC; t += blockDim.x) {
    const int e4 = t & 7;              // which float4 within E
    const int cc = (t >> 3) % 68;
    const int rr = (t >> 3) / 68;
    const int kh = h + rr - kPad;
    const int kw = cc - kPad;
    float* dstp = &kt[((rr * 68) + cc) * kE + e4 * 4];
    if (kh >= 0 && kh < kH && kw >= 0 && kw < kW) {
      const float* srcp = &k[(((long)b * kH + kh) * kW + kw) * kE + e4 * 4];
      ASYNC_COPY16(srcp, dstp);
    } else {
      v4f z = {0.f, 0.f, 0.f, 0.f};
      *(v4f*)dstp = z;
    }
  }
  ASYNC_WAIT();
  __syncthreads();

  const int p  = threadIdx.x;        // 0..255 : (u, ww)
  const int u  = p >> 7;
  const int ww = p & 127;
  const int w  = ww >> 1;

  const float* __restrict__ qp =
      q + (((long)b * kHY + (2 * h + u)) * kWY + ww) * kE;
  float qv[kE];
#pragma unroll
  for (int e = 0; e < kE; ++e) qv[e] = qp[e];

  float lg[25];
  float mx = -1e30f;
#pragma unroll
  for (int dy = 0; dy < kKs; ++dy) {
#pragma unroll
    for (int dx = 0; dx < kKs; ++dx) {
      const float* __restrict__ kp = &kt[((dy * 68) + (w + dx)) * kE];
      float s = 0.f;
#pragma unroll
      for (int e = 0; e < kE; ++e) s = fmaf(qv[e], kp[e], s);
      lg[dy * kKs + dx] = s;
      mx = fmaxf(mx, s);
    }
  }
  float sum = 0.f;
#pragma unroll
  for (int o = 0; o < 25; ++o) { lg[o] = __expf(lg[o] - mx); sum += lg[o]; }
  const float inv = 1.f / sum;

  float* __restrict__ ap = attn + (long)bh * 25 * 256;
#pragma unroll
  for (int o = 0; o < 25; ++o) ap[o * 256 + p] = lg[o] * inv;
}

// ==================================================================
// 5) Aggregation. grid = (B*H, 4 channel-splits of 64). attn weights
//    live in 25 registers/thread. Channels staged 8 at a time: each
//    stage async-copies 8 x 5 x 64 interior floats (16B-aligned rows,
//    stride padded to 72) into LDS; always-zero halo columns are
//    zeroed once up front. 25 FMAs per output, coalesced stores.
// ==================================================================
__global__ __launch_bounds__(256)
void sapa_aggregate_kernel(const float* __restrict__ attn, // (B*H, 25, 256)
                           const float* __restrict__ x,    // (B, C, H, W)
                           float* __restrict__ out) {      // (B, C, 2H, 2W)
  const int bh = blockIdx.x;
  const int b  = bh / kH;
  const int h  = bh % kH;
  const int c0 = blockIdx.y * 64;

  constexpr int CH_TILE = 8;
  // row stride 72: [0..1] unused, [2..3] left zero-halo, [4..67] interior
  // (xw 0..63, starts 16B-aligned), [68..69] right zero-halo, [70..71] unused
  __shared__ __align__(16) float xt[CH_TILE][5 * 72];  // 11520 B

  const int p  = threadIdx.x;
  const int u  = p >> 7;
  const int ww = p & 127;
  const int w  = ww >> 1;

  float aw[25];
  const float* __restrict__ ap = attn + (long)bh * 25 * 256;
#pragma unroll
  for (int o = 0; o < 25; ++o) aw[o] = ap[o * 256 + p];

  // one-time zeroing of the halo columns (never overwritten by stages)
  for (int t = threadIdx.x; t < CH_TILE * 5 * 8; t += blockDim.x) {
    const int eix = t & 7;
    const int rr  = (t >> 3) % 5;
    const int ct  = (t >> 3) / 5;
    const int col = (eix < 4) ? eix : (64 + eix);   // 0..3 and 68..71
    xt[ct][rr * 72 + col] = 0.f;
  }

  for (int cb = 0; cb < 64; cb += CH_TILE) {
    __syncthreads();  // previous stage fully consumed (and halo zeros visible)
    // stage CH_TILE channels: 8 * 5 * 16 aligned float4 transfers
    for (int t = threadIdx.x; t < CH_TILE * 5 * 16; t += blockDim.x) {
      const int q4 = t & 15;            // float4 within row -> xw = q4*4
      const int rr = (t >> 4) % 5;
      const int ct = (t >> 4) / 5;
      const int xh = h + rr - kPad;
      float* dstp = &xt[ct][rr * 72 + 4 + q4 * 4];
      if (xh >= 0 && xh < kH) {
        const float* srcp =
            &x[(((long)b * kC + (c0 + cb + ct)) * kH + xh) * kW + q4 * 4];
        ASYNC_COPY16(srcp, dstp);
      } else {
        v4f z = {0.f, 0.f, 0.f, 0.f};
        *(v4f*)dstp = z;
      }
    }
    // pull the next stage's first rows toward L2 while we wait/compute
    if (cb + CH_TILE < 64)
      __builtin_prefetch(
          &x[(((long)b * kC + (c0 + cb + CH_TILE)) * kH + h) * kW], 0, 0);
    ASYNC_WAIT();
    __syncthreads();

#pragma unroll
    for (int ct = 0; ct < CH_TILE; ++ct) {
      float acc = 0.f;
#pragma unroll
      for (int dy = 0; dy < kKs; ++dy)
#pragma unroll
        for (int dx = 0; dx < kKs; ++dx)
          acc = fmaf(aw[dy * kKs + dx], xt[ct][dy * 72 + (w + dx + 2)], acc);
      out[(((long)b * kC + (c0 + cb + ct)) * kHY + (2 * h + u)) * kWY + ww] = acc;
    }
  }
}

// ==================================================================
// launcher
// ==================================================================
extern "C" void kernel_launch(void* const* d_in, const int* in_sizes, int n_in,
                              void* d_out, int out_size, void* d_ws, size_t ws_size,
                              hipStream_t stream) {
  (void)in_sizes; (void)n_in; (void)out_size; (void)ws_size;
  const float* y      = (const float*)d_in[0];
  const float* x      = (const float*)d_in[1];
  const float* gn_y_w = (const float*)d_in[2];
  const float* gn_y_b = (const float*)d_in[3];
  const float* gn_x_w = (const float*)d_in[4];
  const float* gn_x_b = (const float*)d_in[5];
  const float* q_w    = (const float*)d_in[6];
  const float* q_b    = (const float*)d_in[7];
  const float* k_w    = (const float*)d_in[8];
  const float* k_b    = (const float*)d_in[9];
  float* out = (float*)d_out;
  float* ws  = (float*)d_ws;

  // workspace layout (floats), total ~= 17.1 MB
  float* statsY = ws;                                // 256
  float* statsX = statsY + 2 * kB * kGroups;         // 256
  float* gsY    = statsX + 2 * kB * kGroups;         // 1024
  float* bsY    = gsY + kB * kC;                     // 1024
  float* gsX    = bsY + kB * kC;                     // 1024
  float* bsX    = gsX + kB * kC;                     // 1024
  float* qbuf   = bsX + kB * kC;                     // 4*128*128*32
  float* kbuf   = qbuf + (long)kB * kHY * kWY * kE;  // 4*64*64*32
  float* abuf   = kbuf + (long)kB * kH * kW * kE;    // 4*64*25*256

  // 1) GN stats
  gn_stats_kernel<kHY * kWY><<<kB * kGroups, 256, 0, stream>>>(y, statsY);
  gn_stats_kernel<kH * kW><<<kB * kGroups, 256, 0, stream>>>(x, statsX);

  // 2) fused scale/bias
  gn_fuse_kernel<<<(kB * kC + 255) / 256, 256, 0, stream>>>(statsY, gn_y_w, gn_y_b, gsY, bsY);
  gn_fuse_kernel<<<(kB * kC + 255) / 256, 256, 0, stream>>>(statsX, gn_x_w, gn_x_b, gsX, bsX);

  // 3) WMMA projections (4 waves = 4 row-tiles per 128-thread block)
  {
    const int rowTilesQ = kB * kHY * kWY / 16;  // 4096 waves
    project_wmma_kernel<kHY * kWY><<<rowTilesQ / 4, 128, 0, stream>>>(
        y, gsY, bsY, q_w, q_b, qbuf);
    const int rowTilesK = kB * kH * kW / 16;    // 1024 waves
    project_wmma_kernel<kH * kW><<<rowTilesK / 4, 128, 0, stream>>>(
        x, gsX, bsX, k_w, k_b, kbuf);
  }

  // 4) logits + softmax
  sapa_logits_kernel<<<kB * kH, 256, 0, stream>>>(qbuf, kbuf, abuf);

  // 5) aggregation
  sapa_aggregate_kernel<<<dim3(kB * kH, 4), 256, 0, stream>>>(abuf, x, out);
}
